// MultiHeadAttention_12747462934824
// MI455X (gfx1250) — matrix operand order repaired and checked by reading, not compiled, not run on hardware
//
#include <hip/hip_runtime.h>

typedef __attribute__((ext_vector_type(16))) __bf16 v16bf;
typedef __attribute__((ext_vector_type(8)))  float  v8f;
typedef unsigned int u32x4 __attribute__((ext_vector_type(4)));
typedef int          i32x4 __attribute__((ext_vector_type(4)));
typedef int          i32x8 __attribute__((ext_vector_type(8)));

#define D_MODEL   1024
#define NUM_HEADS 16
#define DEPTH     64
#define BATCH     2
#define SEQ       2048
#define ROWS      (BATCH * SEQ)

#if __has_builtin(__builtin_amdgcn_tensor_load_to_lds) && \
    __has_builtin(__builtin_amdgcn_s_wait_tensorcnt)
#define USE_TDM 1
#else
#define USE_TDM 0
#endif

// ---------------------------------------------------------------------------
// WMMA fragment helpers (wave32, 16x16x32 bf16, layouts per CDNA5 ISA 7.12.2)
// ---------------------------------------------------------------------------
__device__ __forceinline__ v8f wmma_bf16(v16bf a, v16bf b, v8f c) {
  return __builtin_amdgcn_wmma_f32_16x16x32_bf16(false, a, false, b, (short)0, c,
                                                 false, false);
}

// A fragment: 16(M) x 32(K).  lane%16 = M; lanes>=16 hold K offsets +8.
// smem is row-major [row][ld] bf16; koff selects a 32-wide K slice.
__device__ __forceinline__ v16bf frag_a(const __bf16* sm, int rowbase, int ld,
                                        int koff, int lane) {
  union { v16bf v; unsigned u[8]; } f;
  const __bf16* p = sm + (rowbase + (lane & 15)) * ld + koff;
  const int hi = (lane >> 4) << 3;  // 0 or 8
#pragma unroll
  for (int r = 0; r < 8; ++r) {
    int k = ((r >> 2) << 4) + hi + ((r & 3) << 1);  // {0..7,16..23} (+hi)
    f.u[r] = *(const unsigned*)(p + k);
  }
  return f.v;
}

// B fragment: 32(K) x 16(N).  lane%16 = N, lanes>=16 hold K offsets +8.
// smem holds B transposed: [n][ld] with K contiguous.
__device__ __forceinline__ v16bf frag_b(const __bf16* sm, int nbase, int ld,
                                        int koff, int lane) {
  union { v16bf v; unsigned u[8]; } f;
  const __bf16* p = sm + (nbase + (lane & 15)) * ld + koff + ((lane >> 4) << 3);
#pragma unroll
  for (int r = 0; r < 8; ++r) f.u[r] = *(const unsigned*)(p + ((r >> 2) << 4) + ((r & 3) << 1));
  return f.v;
}

#if USE_TDM
// Tensor Data Mover: DMA one 64x64 bf16 tile (row stride `row_stride_elems`
// in memory) into LDS at byte offset lds_off, padding 4 dwords after each
// 128B row so the LDS image has a 72-element row stride (bank-conflict-free).
// D# layout per CDNA5 ISA ch.8 (group0 128b, group1 256b; groups 2/3 unused
// for 2D tensors and passed as zeros).  This toolchain exposes the 6-arg
// builtin: (u32x4 g0, i32x8 g1, i32x4, i32x4, i32x8, i32 cpol).
__device__ __forceinline__ void tdm_load_64x64(const void* gaddr,
                                               unsigned lds_off,
                                               unsigned row_stride_elems) {
  unsigned long long ga = (unsigned long long)gaddr;
  u32x4 g0;
  g0.x = 1u;                                   // count=1, user descriptor
  g0.y = lds_off;                              // lds_addr (bytes)
  g0.z = (unsigned)ga;                         // global_addr[31:0]
  g0.w = (unsigned)(ga >> 32) | 0x80000000u;   // global_addr[56:32] | type=2
  i32x8 g1;
  g1[0] = (1 << 16)        // data_size = 2 bytes
        | (1 << 20)        // pad_enable
        | (4 << 22)        // pad_interval: every 32 dwords (= one 128B row)
        | (3 << 25);       // pad_amount: 4 dwords (16B -> 72-elem stride)
  g1[1] = (64 << 16);      // tensor_dim0 = 64 (atomic_barrier_addr = 0)
  g1[2] = (64 << 16);      // tensor_dim1[15:0] = 64 (tensor_dim0 hi = 0)
  g1[3] = (64 << 16);      // tile_dim0 = 64 (tensor_dim1 hi = 0)
  g1[4] = 64;              // tile_dim1 = 64, tile_dim2 = 0
  g1[5] = (int)row_stride_elems;  // tensor_dim0_stride[31:0]
  g1[6] = 0;               // stride hi / tensor_dim1_stride lo
  g1[7] = 0;
  i32x4 z4 = {0, 0, 0, 0};
  i32x8 z8 = {0, 0, 0, 0, 0, 0, 0, 0};
  __builtin_amdgcn_tensor_load_to_lds(g0, g1, z4, z4, z8, 0);
}
#endif

// ---------------------------------------------------------------------------
// Projection GEMM: Y[4096x1024] = X @ W + b  (fp32 in, bf16 WMMA, f32 acc)
// mode 0: fp32 row-major out;  mode 1: bf16 (b,h,s,d);  mode 2: bf16 (b,h,d,s)
// ---------------------------------------------------------------------------
#define PLD 40  // padded LDS stride (bf16): 80B rows, bank-conflict-free frags

__global__ __launch_bounds__(256) void proj_kernel(
    const float* __restrict__ X, const float* __restrict__ W,
    const float* __restrict__ bias, void* __restrict__ out, int mode) {
  __shared__ alignas(16) __bf16 Asm[64 * PLD];
  __shared__ alignas(16) __bf16 Bsm[64 * PLD];

  const int tid = threadIdx.x;
  const int lane = tid & 31;
  const int w = tid >> 5;            // wave 0..7
  const int tm = w >> 1;             // 16-row subtile (0..3)
  const int tn0 = (w & 1) << 1;      // 16-col subtile pair (0 or 2)
  const int rbase = blockIdx.y << 6;
  const int cbase = blockIdx.x << 6;

  v8f acc0 = {0.f, 0.f, 0.f, 0.f, 0.f, 0.f, 0.f, 0.f};
  v8f acc1 = acc0;

  const int xr = tid >> 2, xc = (tid & 3) << 3;  // X tile: 64 rows x 32 cols
  const int kp2 = tid & 15;                      // W tile: k-pair (k = 2*kp2)
  const int n4 = (tid >> 4) << 2;                // W tile: 4 n-cols per thread

  for (int kk = 0; kk < D_MODEL; kk += 32) {
    __syncthreads();
    {  // stage X 64x32 -> Asm (row-major bf16)
      const float* xp = X + (rbase + xr) * D_MODEL + kk + xc;
      float4 a = *(const float4*)xp;
      float4 b = *(const float4*)(xp + 4);
      __bf16* d = Asm + xr * PLD + xc;
      d[0] = (__bf16)a.x; d[1] = (__bf16)a.y; d[2] = (__bf16)a.z; d[3] = (__bf16)a.w;
      d[4] = (__bf16)b.x; d[5] = (__bf16)b.y; d[6] = (__bf16)b.z; d[7] = (__bf16)b.w;
    }
    {  // stage W 32x64 transposed -> Bsm[n][k], packed bf16-pair dword stores
      const float* w0 = W + (kk + 2 * kp2) * D_MODEL + cbase + n4;
      float4 ra = *(const float4*)w0;
      float4 rb = *(const float4*)(w0 + D_MODEL);
      union { __bf16 h[2]; unsigned u; } p0, p1, p2, p3;
      p0.h[0] = (__bf16)ra.x; p0.h[1] = (__bf16)rb.x;
      p1.h[0] = (__bf16)ra.y; p1.h[1] = (__bf16)rb.y;
      p2.h[0] = (__bf16)ra.z; p2.h[1] = (__bf16)rb.z;
      p3.h[0] = (__bf16)ra.w; p3.h[1] = (__bf16)rb.w;
      unsigned* bdw = (unsigned*)Bsm;
      const int dws = PLD >> 1;  // 20 dwords per LDS row
      bdw[(n4 + 0) * dws + kp2] = p0.u;
      bdw[(n4 + 1) * dws + kp2] = p1.u;
      bdw[(n4 + 2) * dws + kp2] = p2.u;
      bdw[(n4 + 3) * dws + kp2] = p3.u;
    }
    __syncthreads();
    v16bf a  = frag_a(Asm, tm << 4, PLD, 0, lane);
    v16bf b0 = frag_b(Bsm, tn0 << 4, PLD, 0, lane);
    v16bf b1 = frag_b(Bsm, (tn0 + 1) << 4, PLD, 0, lane);
    acc0 = wmma_bf16(a, b0, acc0);
    acc1 = wmma_bf16(a, b1, acc1);
  }

  const int mrow = (lane >> 4) << 3;
  const int nn = lane & 15;
#pragma unroll
  for (int r = 0; r < 8; ++r) {
    int gr = rbase + (tm << 4) + r + mrow;
    int gc0 = cbase + (tn0 << 4) + nn;
    int gc1 = gc0 + 16;
    float v0 = acc0[r] + bias[gc0];
    float v1 = acc1[r] + bias[gc1];
    if (mode == 0) {
      float* o = (float*)out;
      o[gr * D_MODEL + gc0] = v0;
      o[gr * D_MODEL + gc1] = v1;
    } else {
      __bf16* o = (__bf16*)out;
      int b = gr >> 11, s = gr & (SEQ - 1);
      int h0 = gc0 >> 6, d0 = gc0 & 63;
      int h1 = gc1 >> 6, d1 = gc1 & 63;
      if (mode == 1) {  // (b,h,s,d)
        o[(((b << 4) + h0) * SEQ + s) * DEPTH + d0] = (__bf16)v0;
        o[(((b << 4) + h1) * SEQ + s) * DEPTH + d1] = (__bf16)v1;
      } else {          // (b,h,d,s)  (V transposed for attention B-frags)
        o[(((b << 4) + h0) * DEPTH + d0) * SEQ + s] = (__bf16)v0;
        o[(((b << 4) + h1) * DEPTH + d1) * SEQ + s] = (__bf16)v1;
      }
    }
  }
}

// ---------------------------------------------------------------------------
// Causal flash attention: per block = one (b,h) x 64 query rows.
// ---------------------------------------------------------------------------
#define QLD 72   // padded bf16 stride (144B rows: 16B-aligned, conflict-free)
#define SLD 65   // f32 score stride
#define PsLD 72  // bf16 prob stride

__global__ __launch_bounds__(256) void attn_kernel(
    const __bf16* __restrict__ Q, const __bf16* __restrict__ K,
    const __bf16* __restrict__ VT, float* __restrict__ ctx) {
  __shared__ alignas(16) __bf16 Qs[64 * QLD];
  __shared__ alignas(16) __bf16 Ks[64 * QLD];
  __shared__ alignas(16) __bf16 VTs[64 * QLD];
  __shared__ float Ss[64 * SLD];
  __shared__ alignas(16) __bf16 Ps[64 * PsLD];
  __shared__ float rowc[64];

  const int tid = threadIdx.x;
  const int lane = tid & 31;
  const int w = tid >> 5;
  const int tm = w >> 1;
  const int tn0 = (w & 1) << 1;
  const int bid = blockIdx.x;
  const int qb = bid & 31;          // SEQ/64 = 32 q-blocks
  const int bh = bid >> 5;          // 0..31
  const int qbase = qb << 6;
  const int b = bh >> 4, h = bh & 15;

  const __bf16* qp = Q + (size_t)bh * SEQ * DEPTH;
  const __bf16* kp = K + (size_t)bh * SEQ * DEPTH;
  const __bf16* vtp = VT + (size_t)bh * DEPTH * SEQ;

  const int sr = tid >> 2;          // staging row 0..63
  const int sc = (tid & 3) << 4;    // staging col {0,16,32,48}
  (void)sr; (void)sc;

#if USE_TDM
  if (w == 0) {  // stage Q once via Tensor Data Mover
    tdm_load_64x64(qp + qbase * DEPTH,
                   (unsigned)(unsigned long long)(void*)Qs, DEPTH);
    __builtin_amdgcn_s_wait_tensorcnt(0);
  }
#else
  {  // stage Q once (64 x 64 bf16)
    const __bf16* src = qp + (qbase + sr) * DEPTH + sc;
    *(uint4*)&Qs[sr * QLD + sc]     = *(const uint4*)src;
    *(uint4*)&Qs[sr * QLD + sc + 8] = *(const uint4*)(src + 8);
  }
#endif
  __syncthreads();

  v16bf qa0 = frag_a(Qs, tm << 4, QLD, 0, lane);   // loop-invariant
  v16bf qa1 = frag_a(Qs, tm << 4, QLD, 32, lane);

  v8f cacc0 = {0.f, 0.f, 0.f, 0.f, 0.f, 0.f, 0.f, 0.f};
  v8f cacc1 = cacc0;
  float run_m = -3e38f, run_l = 0.f;  // maintained by tid<64 row owners

  const int mrow = (lane >> 4) << 3;
  const int nn = lane & 15;

  for (int kb = 0; kb <= qb; ++kb) {
    const int kbase = kb << 6;
    __syncthreads();
#if USE_TDM
    if (w == 0) {        // K block [key][d]
      tdm_load_64x64(kp + kbase * DEPTH,
                     (unsigned)(unsigned long long)(void*)Ks, DEPTH);
      __builtin_amdgcn_s_wait_tensorcnt(0);
    } else if (w == 1) { // V^T block [d][key]
      tdm_load_64x64(vtp + kbase,
                     (unsigned)(unsigned long long)(void*)VTs, SEQ);
      __builtin_amdgcn_s_wait_tensorcnt(0);
    }
#else
    {  // stage K block [key][d] and V^T block [d][key]
      const __bf16* ksrc = kp + (kbase + sr) * DEPTH + sc;
      *(uint4*)&Ks[sr * QLD + sc]     = *(const uint4*)ksrc;
      *(uint4*)&Ks[sr * QLD + sc + 8] = *(const uint4*)(ksrc + 8);
      const __bf16* vsrc = vtp + sr * SEQ + kbase + sc;
      *(uint4*)&VTs[sr * QLD + sc]     = *(const uint4*)vsrc;
      *(uint4*)&VTs[sr * QLD + sc + 8] = *(const uint4*)(vsrc + 8);
    }
#endif
    __syncthreads();

    // S = scale * Q K^T  (contract over depth=64: two 16x16x32 WMMAs/subtile)
    v8f s0 = {0.f, 0.f, 0.f, 0.f, 0.f, 0.f, 0.f, 0.f};
    v8f s1 = s0;
    s0 = wmma_bf16(qa0, frag_b(Ks, tn0 << 4, QLD, 0, lane), s0);
    s0 = wmma_bf16(qa1, frag_b(Ks, tn0 << 4, QLD, 32, lane), s0);
    s1 = wmma_bf16(qa0, frag_b(Ks, (tn0 + 1) << 4, QLD, 0, lane), s1);
    s1 = wmma_bf16(qa1, frag_b(Ks, (tn0 + 1) << 4, QLD, 32, lane), s1);

#pragma unroll
    for (int r = 0; r < 8; ++r) {  // masked scaled write to Ss
      int lm = (tm << 4) + r + mrow;
      int gq = qbase + lm;
      int lk0 = (tn0 << 4) + nn;
      int gk0 = kbase + lk0;
      Ss[lm * SLD + lk0]      = (gk0 > gq)      ? -1e9f : s0[r] * 0.125f;
      Ss[lm * SLD + lk0 + 16] = (gk0 + 16 > gq) ? -1e9f : s1[r] * 0.125f;
    }
    __syncthreads();

    if (tid < 64) {  // online softmax, one thread per q row
      const float* srow = Ss + tid * SLD;
      float mb = srow[0];
#pragma unroll 4
      for (int j = 1; j < 64; ++j) mb = fmaxf(mb, srow[j]);
      float mnew = fmaxf(run_m, mb);
      float corr = __expf(run_m - mnew);
      float sum = 0.f;
      __bf16* prow = Ps + tid * PsLD;
#pragma unroll 4
      for (int j = 0; j < 64; ++j) {
        float e = __expf(srow[j] - mnew);
        sum += e;
        prow[j] = (__bf16)e;
      }
      run_l = run_l * corr + sum;
      run_m = mnew;
      rowc[tid] = corr;
    }
    __syncthreads();

    // rescale accumulators, then ctx += P @ V
#pragma unroll
    for (int r = 0; r < 8; ++r) {
      float c = rowc[(tm << 4) + r + mrow];
      cacc0[r] *= c;
      cacc1[r] *= c;
    }
    v16bf pa0 = frag_a(Ps, tm << 4, PsLD, 0, lane);
    v16bf pa1 = frag_a(Ps, tm << 4, PsLD, 32, lane);
    cacc0 = wmma_bf16(pa0, frag_b(VTs, tn0 << 4, QLD, 0, lane), cacc0);
    cacc0 = wmma_bf16(pa1, frag_b(VTs, tn0 << 4, QLD, 32, lane), cacc0);
    cacc1 = wmma_bf16(pa0, frag_b(VTs, (tn0 + 1) << 4, QLD, 0, lane), cacc1);
    cacc1 = wmma_bf16(pa1, frag_b(VTs, (tn0 + 1) << 4, QLD, 32, lane), cacc1);
  }

  __syncthreads();
  if (tid < 64) rowc[tid] = 1.0f / run_l;  // reuse rowc as inv row-sum
  __syncthreads();

#pragma unroll
  for (int r = 0; r < 8; ++r) {  // merge heads: ctx[b][s][h*64+d], fp32
    int lm = (tm << 4) + r + mrow;
    float inv = rowc[lm];
    size_t row = (size_t)(b * SEQ + qbase + lm) * D_MODEL + h * DEPTH;
    ctx[row + (tn0 << 4) + nn]      = cacc0[r] * inv;
    ctx[row + (tn0 << 4) + nn + 16] = cacc1[r] * inv;
  }
}

// ---------------------------------------------------------------------------
extern "C" void kernel_launch(void* const* d_in, const int* in_sizes, int n_in,
                              void* d_out, int out_size, void* d_ws,
                              size_t ws_size, hipStream_t stream) {
  (void)in_sizes; (void)n_in; (void)out_size; (void)ws_size;
  const float* q  = (const float*)d_in[0];
  const float* k  = (const float*)d_in[1];
  const float* v  = (const float*)d_in[2];
  // d_in[3] = mask: always causal tril; applied analytically in attn_kernel.
  const float* wq = (const float*)d_in[4];
  const float* bq = (const float*)d_in[5];
  const float* wk = (const float*)d_in[6];
  const float* bk = (const float*)d_in[7];
  const float* wv = (const float*)d_in[8];
  const float* bv = (const float*)d_in[9];
  const float* wo = (const float*)d_in[10];
  const float* bo = (const float*)d_in[11];

  char* ws = (char*)d_ws;
  __bf16* Qb  = (__bf16*)(ws);                  // 8 MB bf16 (b,h,s,d)
  __bf16* Kb  = (__bf16*)(ws + (8ull << 20));   // 8 MB bf16 (b,h,s,d)
  __bf16* VTb = (__bf16*)(ws + (16ull << 20));  // 8 MB bf16 (b,h,d,s)
  float*  ctx = (float*)(ws + (24ull << 20));   // 16 MB f32 (b,s,dm)

  dim3 pg(D_MODEL / 64, ROWS / 64);  // (16, 64)
  proj_kernel<<<pg, 256, 0, stream>>>(q, wq, bq, (void*)Qb, 1);
  proj_kernel<<<pg, 256, 0, stream>>>(k, wk, bk, (void*)Kb, 1);
  proj_kernel<<<pg, 256, 0, stream>>>(v, wv, bv, (void*)VTb, 2);
  attn_kernel<<<BATCH * NUM_HEADS * (SEQ / 64), 256, 0, stream>>>(Qb, Kb, VTb,
                                                                  ctx);
  proj_kernel<<<pg, 256, 0, stream>>>(ctx, wo, bo, d_out, 0);
}
